// OuterModel_29051158790490
// MI455X (gfx1250) — matrix operand order repaired and checked by reading, not compiled, not run on hardware
//
#include <hip/hip_runtime.h>
#include <hip/hip_bf16.h>

typedef __attribute__((ext_vector_type(16))) _Float16 v16h;
typedef __attribute__((ext_vector_type(8)))  _Float16 v8h;
typedef __attribute__((ext_vector_type(8)))  float    v8f;

union F16Frag { v16h v; v8h h[2]; };

// Problem constants
constexpr int Bb    = 8;
constexpr int Ll    = 2048;
constexpr int Dd    = 1024;
constexpr int Mm    = Bb * Ll;        // 16384 GEMM rows
constexpr int NCH   = 32;             // scan chunks
constexpr int CHL   = Ll / NCH;       // 64 steps per chunk
constexpr size_t NT = (size_t)Bb * Ll * Dd;

// ---------------------------------------------------------------------------
// CDNA5 async global->LDS staging (ASYNCcnt path), guarded so the file
// compiles on toolchains that do not declare the builtin.
// Probe-derived signature: (AS1 int4* src, AS3 int4* dst, int offset, int cpol)
// ---------------------------------------------------------------------------
#if defined(__AMDGCN__) && __has_builtin(__builtin_amdgcn_global_load_async_to_lds_b128)
#define ASYNC_STAGE 1
#else
#define ASYNC_STAGE 0
#endif

#if ASYNC_STAGE
typedef int v4i32 __attribute__((vector_size(16)));
typedef __attribute__((address_space(1))) v4i32* gq_t;
typedef __attribute__((address_space(3))) v4i32* lq_t;
__device__ __forceinline__ void g2l_async16(const _Float16* g, _Float16* l) {
  __builtin_amdgcn_global_load_async_to_lds_b128((gq_t)g, (lq_t)l, 0, 0);
}
__device__ __forceinline__ void async_wait0() {
#if __has_builtin(__builtin_amdgcn_s_wait_asynccnt)
  __builtin_amdgcn_s_wait_asynccnt(0);
#else
  asm volatile("s_wait_asynccnt 0x0" ::: "memory");
#endif
}
#endif

// ---------------------------------------------------------------------------
// Weight prep: f32 -> f16 copies of wr/wi/wo ; l2a[d] = log2(sigmoid(log_a[d]))
// ---------------------------------------------------------------------------
__global__ __launch_bounds__(256) void cvt_weights_kernel(
    const float* __restrict__ wr, const float* __restrict__ wi,
    const float* __restrict__ wo, const float* __restrict__ la,
    _Float16* __restrict__ wr16, _Float16* __restrict__ wi16,
    _Float16* __restrict__ wo16, float* __restrict__ l2a,
    int nW, int nA) {
  int i = blockIdx.x * 256 + threadIdx.x;
  if (i < nW) {
    wr16[i] = (_Float16)wr[i];
    wi16[i] = (_Float16)wi[i];
    wo16[i] = (_Float16)wo[i];
  }
  if (i < nA) {
    float s = 1.f / (1.f + __expf(-la[i]));
    l2a[i] = __log2f(s);
  }
}

// ---------------------------------------------------------------------------
// Causal depthwise conv1d (K=4, left pad 3): xc = cb + sum_k x[l-3+k]*cw[k]
// grid (D/256, L, B)
// ---------------------------------------------------------------------------
__global__ __launch_bounds__(256) void conv_kernel(
    const float* __restrict__ X, const float* __restrict__ cw,
    const float* __restrict__ cb, float* __restrict__ xc32,
    _Float16* __restrict__ xc16) {
  const int d = blockIdx.x * 256 + threadIdx.x;
  const int l = blockIdx.y, b = blockIdx.z;
  const float4 w = *(const float4*)(cw + d * 4);
  float acc = cb[d];
  const size_t rb = ((size_t)b * Ll + l) * Dd + d;
  if (l >= 3) acc += X[rb - 3 * (size_t)Dd] * w.x;
  if (l >= 2) acc += X[rb - 2 * (size_t)Dd] * w.y;
  if (l >= 1) acc += X[rb - 1 * (size_t)Dd] * w.z;
  acc += X[rb] * w.w;
  xc32[rb] = acc;
  xc16[rb] = (_Float16)acc;
}

// ---------------------------------------------------------------------------
// Dual GEMM (r and i gates share the A operand) + fused gate epilogue.
//   Rpre = A16 @ Wr^T, Ipre = A16 @ Wi^T    (A16:[M,K], W:[N,K], K=N=1024)
//   r=sigm(Rpre+wrb); i=sigm(Ipre+wib); p=8*r*log2(a_base);
//   a=exp2(p); store LG = max(p*ln2, ln 1e-7)  (== log(max(a,1e-7)))
//   b=sqrt(max(1-a^2,1e-6))*i*xc
// Block: 256 thr (8 waves), tile 64x128, wave tile 32x32 per GEMM.
// ---------------------------------------------------------------------------
__global__ __launch_bounds__(256) void gemm_gates_kernel(
    const _Float16* __restrict__ A16, const _Float16* __restrict__ Wr,
    const _Float16* __restrict__ Wi, const float* __restrict__ wrb,
    const float* __restrict__ wib, const float* __restrict__ l2a,
    const float* __restrict__ xc32, float* __restrict__ LGout,
    float* __restrict__ Bout) {
  const int m0 = blockIdx.x * 64;
  const int n0 = blockIdx.y * 128;
  const int t = threadIdx.x;
  const int wv = t >> 5, lane = t & 31;
  const int wm = wv & 1, wn = wv >> 1;
  const int lr = lane & 15, lh = lane >> 4;

  __shared__ alignas(16) _Float16 AsL[2][64 * 32];
  __shared__ alignas(16) _Float16 BrL[2][128 * 32];
  __shared__ alignas(16) _Float16 BiL[2][128 * 32];

  v8f accR[2][2] = {};
  v8f accI[2][2] = {};

  // Staging index math (one 16B segment for A, two each for Wr/Wi per thread)
  const int srow = t >> 2;
  const int ssg = (t & 3) * 8;
  const size_t aG = (size_t)(m0 + srow) * Dd + ssg;
  const size_t b0G = (size_t)(n0 + srow) * Dd + ssg;
  const size_t b1G = (size_t)(n0 + 64 + srow) * Dd + ssg;
  const int aL = srow * 32 + ssg;
  const int b0L = srow * 32 + ssg;
  const int b1L = (64 + srow) * 32 + ssg;

  // Fragment read bases (A: K-split 16/16 per lane-half; B: K-split 16/16)
  int aF[2], bF[2];
#pragma unroll
  for (int m = 0; m < 2; ++m) aF[m] = (wm * 32 + m * 16 + lr) * 32 + lh * 16;
#pragma unroll
  for (int n = 0; n < 2; ++n) bF[n] = (wn * 32 + n * 16 + lr) * 32 + lh * 16;

#if ASYNC_STAGE
  g2l_async16(A16 + aG, &AsL[0][aL]);
  g2l_async16(Wr + b0G, &BrL[0][b0L]);
  g2l_async16(Wr + b1G, &BrL[0][b1L]);
  g2l_async16(Wi + b0G, &BiL[0][b0L]);
  g2l_async16(Wi + b1G, &BiL[0][b1L]);
  async_wait0();
  __syncthreads();
#else
  v8h ra, rr0, rr1, ri0, ri1;
  ra = *(const v8h*)(A16 + aG);
  rr0 = *(const v8h*)(Wr + b0G);
  rr1 = *(const v8h*)(Wr + b1G);
  ri0 = *(const v8h*)(Wi + b0G);
  ri1 = *(const v8h*)(Wi + b1G);
  *(v8h*)&AsL[0][aL] = ra;
  *(v8h*)&BrL[0][b0L] = rr0;
  *(v8h*)&BrL[0][b1L] = rr1;
  *(v8h*)&BiL[0][b0L] = ri0;
  *(v8h*)&BiL[0][b1L] = ri1;
  __syncthreads();
#endif

  constexpr int KS = Dd / 32;
  for (int ks = 0; ks < KS; ++ks) {
    const int cur = ks & 1;
    const int nxt = cur ^ 1;
    const bool more = (ks + 1) < KS;
#if ASYNC_STAGE
    if (more) {
      const size_t ko = (size_t)(ks + 1) * 32;
      g2l_async16(A16 + aG + ko, &AsL[nxt][aL]);
      g2l_async16(Wr + b0G + ko, &BrL[nxt][b0L]);
      g2l_async16(Wr + b1G + ko, &BrL[nxt][b1L]);
      g2l_async16(Wi + b0G + ko, &BiL[nxt][b0L]);
      g2l_async16(Wi + b1G + ko, &BiL[nxt][b1L]);
    }
#else
    if (more) {
      const size_t ko = (size_t)(ks + 1) * 32;
      ra = *(const v8h*)(A16 + aG + ko);
      rr0 = *(const v8h*)(Wr + b0G + ko);
      rr1 = *(const v8h*)(Wr + b1G + ko);
      ri0 = *(const v8h*)(Wi + b0G + ko);
      ri1 = *(const v8h*)(Wi + b1G + ko);
    }
#endif
    F16Frag fa[2], fr[2], fi[2];
#pragma unroll
    for (int m = 0; m < 2; ++m) {
      fa[m].h[0] = *(const v8h*)&AsL[cur][aF[m]];
      fa[m].h[1] = *(const v8h*)&AsL[cur][aF[m] + 8];
    }
#pragma unroll
    for (int n = 0; n < 2; ++n) {
      fr[n].h[0] = *(const v8h*)&BrL[cur][bF[n]];
      fr[n].h[1] = *(const v8h*)&BrL[cur][bF[n] + 8];
      fi[n].h[0] = *(const v8h*)&BiL[cur][bF[n]];
      fi[n].h[1] = *(const v8h*)&BiL[cur][bF[n] + 8];
    }
#pragma unroll
    for (int m = 0; m < 2; ++m)
#pragma unroll
      for (int n = 0; n < 2; ++n) {
        accR[m][n] = __builtin_amdgcn_wmma_f32_16x16x32_f16(
            false, fa[m].v, false, fr[n].v, (short)0, accR[m][n], false, false);
        accI[m][n] = __builtin_amdgcn_wmma_f32_16x16x32_f16(
            false, fa[m].v, false, fi[n].v, (short)0, accI[m][n], false, false);
      }
#if ASYNC_STAGE
    if (more) {
      async_wait0();
      __syncthreads();
    }
#else
    if (more) {
      __syncthreads();
      *(v8h*)&AsL[nxt][aL] = ra;
      *(v8h*)&BrL[nxt][b0L] = rr0;
      *(v8h*)&BrL[nxt][b1L] = rr1;
      *(v8h*)&BiL[nxt][b0L] = ri0;
      *(v8h*)&BiL[nxt][b1L] = ri1;
      __syncthreads();
    }
#endif
  }

  // Fused gate epilogue. C/D layout: VGPR r -> row r + 8*lh, lane -> col lr.
#pragma unroll
  for (int n = 0; n < 2; ++n) {
    const int col = n0 + wn * 32 + n * 16 + lr;
    const float wrbv = wrb[col];
    const float wibv = wib[col];
    const float l2av = l2a[col];
#pragma unroll
    for (int m = 0; m < 2; ++m) {
      const int rowBase = m0 + wm * 32 + m * 16 + lh * 8;
#pragma unroll
      for (int r = 0; r < 8; ++r) {
        const size_t idx = (size_t)(rowBase + r) * Dd + col;
        const float rp = accR[m][n][r] + wrbv;
        const float ip = accI[m][n][r] + wibv;
        const float rg = 1.f / (1.f + __expf(-rp));
        const float ig = 1.f / (1.f + __expf(-ip));
        const float p8 = 8.f * rg * l2av;       // log2(a)
        const float at = exp2f(p8);
        const float xc = xc32[idx];
        const float om = fmaxf(1.f - at * at, 1e-6f);
        const float bt = sqrtf(om) * ig * xc;
        // log(max(a,1e-7)) == max(ln a, ln 1e-7) by monotonicity
        LGout[idx] = fmaxf(p8 * 0.6931471805599453f, -16.11809565095832f);
        Bout[idx] = bt;
      }
    }
  }
}

// ---------------------------------------------------------------------------
// Single GEMM: Y = H16 @ Wo^T  (f32 out)
// ---------------------------------------------------------------------------
__global__ __launch_bounds__(256) void gemm_wo_kernel(
    const _Float16* __restrict__ A16, const _Float16* __restrict__ Wo,
    float* __restrict__ Y) {
  const int m0 = blockIdx.x * 64;
  const int n0 = blockIdx.y * 128;
  const int t = threadIdx.x;
  const int wv = t >> 5, lane = t & 31;
  const int wm = wv & 1, wn = wv >> 1;
  const int lr = lane & 15, lh = lane >> 4;

  __shared__ alignas(16) _Float16 AsL[2][64 * 32];
  __shared__ alignas(16) _Float16 BoL[2][128 * 32];

  v8f acc[2][2] = {};

  const int srow = t >> 2;
  const int ssg = (t & 3) * 8;
  const size_t aG = (size_t)(m0 + srow) * Dd + ssg;
  const size_t b0G = (size_t)(n0 + srow) * Dd + ssg;
  const size_t b1G = (size_t)(n0 + 64 + srow) * Dd + ssg;
  const int aL = srow * 32 + ssg;
  const int b0L = srow * 32 + ssg;
  const int b1L = (64 + srow) * 32 + ssg;

  int aF[2], bF[2];
#pragma unroll
  for (int m = 0; m < 2; ++m) aF[m] = (wm * 32 + m * 16 + lr) * 32 + lh * 16;
#pragma unroll
  for (int n = 0; n < 2; ++n) bF[n] = (wn * 32 + n * 16 + lr) * 32 + lh * 16;

#if ASYNC_STAGE
  g2l_async16(A16 + aG, &AsL[0][aL]);
  g2l_async16(Wo + b0G, &BoL[0][b0L]);
  g2l_async16(Wo + b1G, &BoL[0][b1L]);
  async_wait0();
  __syncthreads();
#else
  v8h ra, rb0, rb1;
  ra = *(const v8h*)(A16 + aG);
  rb0 = *(const v8h*)(Wo + b0G);
  rb1 = *(const v8h*)(Wo + b1G);
  *(v8h*)&AsL[0][aL] = ra;
  *(v8h*)&BoL[0][b0L] = rb0;
  *(v8h*)&BoL[0][b1L] = rb1;
  __syncthreads();
#endif

  constexpr int KS = Dd / 32;
  for (int ks = 0; ks < KS; ++ks) {
    const int cur = ks & 1;
    const int nxt = cur ^ 1;
    const bool more = (ks + 1) < KS;
#if ASYNC_STAGE
    if (more) {
      const size_t ko = (size_t)(ks + 1) * 32;
      g2l_async16(A16 + aG + ko, &AsL[nxt][aL]);
      g2l_async16(Wo + b0G + ko, &BoL[nxt][b0L]);
      g2l_async16(Wo + b1G + ko, &BoL[nxt][b1L]);
    }
#else
    if (more) {
      const size_t ko = (size_t)(ks + 1) * 32;
      ra = *(const v8h*)(A16 + aG + ko);
      rb0 = *(const v8h*)(Wo + b0G + ko);
      rb1 = *(const v8h*)(Wo + b1G + ko);
    }
#endif
    F16Frag fa[2], fb[2];
#pragma unroll
    for (int m = 0; m < 2; ++m) {
      fa[m].h[0] = *(const v8h*)&AsL[cur][aF[m]];
      fa[m].h[1] = *(const v8h*)&AsL[cur][aF[m] + 8];
    }
#pragma unroll
    for (int n = 0; n < 2; ++n) {
      fb[n].h[0] = *(const v8h*)&BoL[cur][bF[n]];
      fb[n].h[1] = *(const v8h*)&BoL[cur][bF[n] + 8];
    }
#pragma unroll
    for (int m = 0; m < 2; ++m)
#pragma unroll
      for (int n = 0; n < 2; ++n)
        acc[m][n] = __builtin_amdgcn_wmma_f32_16x16x32_f16(
            false, fa[m].v, false, fb[n].v, (short)0, acc[m][n], false, false);
#if ASYNC_STAGE
    if (more) {
      async_wait0();
      __syncthreads();
    }
#else
    if (more) {
      __syncthreads();
      *(v8h*)&AsL[nxt][aL] = ra;
      *(v8h*)&BoL[nxt][b0L] = rb0;
      *(v8h*)&BoL[nxt][b1L] = rb1;
      __syncthreads();
    }
#endif
  }

#pragma unroll
  for (int n = 0; n < 2; ++n) {
    const int col = n0 + wn * 32 + n * 16 + lr;
#pragma unroll
    for (int m = 0; m < 2; ++m) {
      const int rowBase = m0 + wm * 32 + m * 16 + lh * 8;
#pragma unroll
      for (int r = 0; r < 8; ++r)
        Y[(size_t)(rowBase + r) * Dd + col] = acc[m][n][r];
    }
  }
}

// ---------------------------------------------------------------------------
// Chunked log-space scan (matches reference incl. log_A clamp at -80).
// LG already holds log(max(a,1e-7)) from the gate epilogue.
// Pass 1: per-chunk sum of log a.                   grid (D/256, NCH, B)
// ---------------------------------------------------------------------------
__global__ __launch_bounds__(256) void scan_loga_kernel(
    const float* __restrict__ LG, float* __restrict__ lagg) {
  const int d = blockIdx.x * 256 + threadIdx.x;
  const int ch = blockIdx.y, b = blockIdx.z;
  size_t base = ((size_t)b * Ll + (size_t)ch * CHL) * Dd + d;
  float s = 0.f;
  for (int tt = 0; tt < CHL; ++tt) s += LG[base + (size_t)tt * Dd];
  lagg[((size_t)b * NCH + ch) * Dd + d] = s;
}

// Pass 2 (used twice): serial exclusive prefix over NCH chunks. grid (D/256,1,B)
__global__ __launch_bounds__(256) void chunk_prefix_kernel(float* __restrict__ agg) {
  const int d = blockIdx.x * 256 + threadIdx.x;
  const int b = blockIdx.z;
  float run = 0.f;
  for (int ch = 0; ch < NCH; ++ch) {
    const size_t i = ((size_t)b * NCH + ch) * Dd + d;
    const float v = agg[i];
    agg[i] = run;
    run += v;
  }
}

// Pass 3: per-chunk sum of b*exp(-clamped logA), given logA chunk offsets.
__global__ __launch_bounds__(256) void scan_csum_kernel(
    const float* __restrict__ LG, const float* __restrict__ Bv,
    const float* __restrict__ lagg, float* __restrict__ cagg) {
  const int d = blockIdx.x * 256 + threadIdx.x;
  const int ch = blockIdx.y, b = blockIdx.z;
  const size_t ai = ((size_t)b * NCH + ch) * Dd + d;
  size_t base = ((size_t)b * Ll + (size_t)ch * CHL) * Dd + d;
  float la = lagg[ai];
  float s = 0.f;
  for (int tt = 0; tt < CHL; ++tt) {
    const size_t ix = base + (size_t)tt * Dd;
    la += LG[ix];
    const float laC = fmaxf(la, -80.f);
    s += Bv[ix] * __expf(-laC);
  }
  cagg[ai] = s;
}

// Pass 5: final h_t = exp(logA_t) * cumsum_t(b*exp(-logA)), write f16 for GEMM.
__global__ __launch_bounds__(256) void scan_final_kernel(
    const float* __restrict__ LG, const float* __restrict__ Bv,
    const float* __restrict__ lagg, const float* __restrict__ cagg,
    _Float16* __restrict__ H16) {
  const int d = blockIdx.x * 256 + threadIdx.x;
  const int ch = blockIdx.y, b = blockIdx.z;
  const size_t ai = ((size_t)b * NCH + ch) * Dd + d;
  size_t base = ((size_t)b * Ll + (size_t)ch * CHL) * Dd + d;
  float la = lagg[ai];
  float c = cagg[ai];
  for (int tt = 0; tt < CHL; ++tt) {
    const size_t ix = base + (size_t)tt * Dd;
    la += LG[ix];
    const float laC = fmaxf(la, -80.f);
    c += Bv[ix] * __expf(-laC);
    H16[ix] = (_Float16)(__expf(laC) * c);
  }
}

// ---------------------------------------------------------------------------
// RMSNorm per row: out = y * w * rsqrt(mean(y^2)+1e-6). grid: M rows x 256 thr
// ---------------------------------------------------------------------------
__global__ __launch_bounds__(256) void rmsnorm_kernel(
    const float* __restrict__ Y, const float* __restrict__ w,
    float* __restrict__ O) {
  const int row = blockIdx.x;
  const int t = threadIdx.x;
  const size_t base = (size_t)row * Dd + t * 4;
  const float4 v = *(const float4*)(Y + base);
  float ss = v.x * v.x + v.y * v.y + v.z * v.z + v.w * v.w;
#pragma unroll
  for (int o = 16; o > 0; o >>= 1) ss += __shfl_down(ss, o);
  __shared__ float red[8];
  if ((t & 31) == 0) red[t >> 5] = ss;
  __syncthreads();
  if (t == 0) {
    float tot = 0.f;
    for (int i = 0; i < 8; ++i) tot += red[i];
    red[0] = rsqrtf(tot * (1.f / (float)Dd) + 1e-6f);
  }
  __syncthreads();
  const float sc = red[0];
  const float4 wv = *(const float4*)(w + t * 4);
  float4 o;
  o.x = v.x * wv.x * sc;
  o.y = v.y * wv.y * sc;
  o.z = v.z * wv.z * sc;
  o.w = v.w * wv.w * sc;
  *(float4*)(O + base) = o;
}

// ---------------------------------------------------------------------------
extern "C" void kernel_launch(void* const* d_in, const int* in_sizes, int n_in,
                              void* d_out, int out_size, void* d_ws, size_t ws_size,
                              hipStream_t stream) {
  (void)in_sizes; (void)n_in; (void)out_size; (void)ws_size;
  const float* x          = (const float*)d_in[0];
  const float* conv_w     = (const float*)d_in[1];
  const float* conv_b     = (const float*)d_in[2];
  const float* wr_w       = (const float*)d_in[3];
  const float* wr_b       = (const float*)d_in[4];
  const float* wi_w       = (const float*)d_in[5];
  const float* wi_b       = (const float*)d_in[6];
  const float* log_a      = (const float*)d_in[7];
  const float* wo_w       = (const float*)d_in[8];
  const float* norm_w     = (const float*)d_in[9];
  const float* norm_out_w = (const float*)d_in[10];
  float* out = (float*)d_out;

  // Bump allocator over workspace (256B aligned)
  char* ws = (char*)d_ws;
  size_t off = 0;
  auto bump = [&](size_t bytes) -> void* {
    void* p = ws + off;
    off += (bytes + 255) & ~(size_t)255;
    return p;
  };
  const size_t WN = 3ull * Dd * Dd;  // weights per family across layers
  _Float16* wr16 = (_Float16*)bump(WN * 2);
  _Float16* wi16 = (_Float16*)bump(WN * 2);
  _Float16* wo16 = (_Float16*)bump(WN * 2);
  float* l2a     = (float*)bump(3ull * Dd * 4);
  float* Xb      = (float*)bump(NT * 4);
  float* XC32    = (float*)bump(NT * 4);
  _Float16* XC16 = (_Float16*)bump(NT * 2);
  float* LG      = (float*)bump(NT * 4);   // holds log(a_t)
  float* Bv      = (float*)bump(NT * 4);
  _Float16* H16  = (_Float16*)bump(NT * 2);
  float* lagg    = (float*)bump((size_t)Bb * NCH * Dd * 4);
  float* cagg    = (float*)bump((size_t)Bb * NCH * Dd * 4);
  float* Y = XC32;  // XC32 is dead after the gate-GEMM epilogue

  cvt_weights_kernel<<<(int)((WN + 255) / 256), 256, 0, stream>>>(
      wr_w, wi_w, wo_w, log_a, wr16, wi16, wo16, l2a, (int)WN, 3 * Dd);

  const dim3 gridGemm(Mm / 64, Dd / 128);
  const dim3 gridConv(Dd / 256, Ll, Bb);
  const dim3 gridScan(Dd / 256, NCH, Bb);
  const dim3 gridPref(Dd / 256, 1, Bb);

  const float* cur = x;
  for (int l = 0; l < 3; ++l) {
    conv_kernel<<<gridConv, 256, 0, stream>>>(
        cur, conv_w + (size_t)l * Dd * 4, conv_b + (size_t)l * Dd, XC32, XC16);
    gemm_gates_kernel<<<gridGemm, 256, 0, stream>>>(
        XC16, wr16 + (size_t)l * Dd * Dd, wi16 + (size_t)l * Dd * Dd,
        wr_b + (size_t)l * Dd, wi_b + (size_t)l * Dd, l2a + (size_t)l * Dd,
        XC32, LG, Bv);
    scan_loga_kernel<<<gridScan, 256, 0, stream>>>(LG, lagg);
    chunk_prefix_kernel<<<gridPref, 256, 0, stream>>>(lagg);
    scan_csum_kernel<<<gridScan, 256, 0, stream>>>(LG, Bv, lagg, cagg);
    chunk_prefix_kernel<<<gridPref, 256, 0, stream>>>(cagg);
    scan_final_kernel<<<gridScan, 256, 0, stream>>>(LG, Bv, lagg, cagg, H16);
    gemm_wo_kernel<<<gridGemm, 256, 0, stream>>>(
        H16, wo16 + (size_t)l * Dd * Dd, Y);
    rmsnorm_kernel<<<Mm, 256, 0, stream>>>(Y, norm_w + (size_t)l * Dd, Xb);
    cur = Xb;
  }
  rmsnorm_kernel<<<Mm, 256, 0, stream>>>(Xb, norm_out_w, out);
}
